// DCM_23845658427856
// MI455X (gfx1250) — matrix operand order repaired and checked by reading, not compiled, not run on hardware
//
#include <hip/hip_runtime.h>

// ---------------------------------------------------------------------------
// DCM (dynamic conv module) fused pipeline for MI455X (gfx1250, wave32, WMMA)
//   K1: adaptive 3x3 avg pool of x                      (reads x once, 128MB)
//   K2: filter-generating 1x1 conv -> gf[4,512,3,3]     (tiny)
//   K3: xr = relu(redu_w @ x + b)     split-f16 WMMA GEMM (M=512,N=4096,K=2048)x4
//   K4: depthwise 3x3 with gf, relu                     (streaming)
//   K5: out = relu(fusion_w @ out2 + b) split-f16 WMMA GEMM (K=512)
// Split-f16: A=Ah+Al, B=Bh+Bl, D += AhBh + AhBl + AlBh  (~1e-6 rel err vs f32)
// GEMM staging is register-pipelined: tile k+1 global loads are in flight
// while tile k runs its 12 WMMAs from LDS.
// ---------------------------------------------------------------------------

typedef __attribute__((ext_vector_type(16))) _Float16 v16h;
typedef __attribute__((ext_vector_type(8)))  _Float16 h8;
typedef __attribute__((ext_vector_type(4)))  _Float16 h4;
typedef __attribute__((ext_vector_type(8)))  float    v8f;

#define NB   4
#define CIN  2048
#define HW   4096
#define CMID 512

#define CAT16(lo, hi) __builtin_shufflevector((lo), (hi), 0,1,2,3,4,5,6,7,8,9,10,11,12,13,14,15)

// ---------------- K1: adaptive avg pool 64x64 -> 3x3 (bins 22 wide, overlap)
__global__ __launch_bounds__(256) void dcm_pool(const float* __restrict__ x,
                                                float* __restrict__ pooled) {
  const int bc = blockIdx.x;                  // b*CIN + cin
  const float* xp = x + (size_t)bc * HW;
  float acc[9];
#pragma unroll
  for (int j = 0; j < 9; ++j) acc[j] = 0.f;
#pragma unroll
  for (int r = 0; r < 16; ++r) {
    const int idx = threadIdx.x + (r << 8);
    const float v = xp[idx];
    const int h = idx >> 6, w = idx & 63;
#pragma unroll
    for (int p = 0; p < 3; ++p) {
      const int hs = (p * 64) / 3;            // 0,21,42
      const int he = ((p + 1) * 64 + 2) / 3;  // 22,43,64
      if (h >= hs && h < he) {
#pragma unroll
        for (int q = 0; q < 3; ++q) {
          const int ws_ = (q * 64) / 3;
          const int we  = ((q + 1) * 64 + 2) / 3;
          if (w >= ws_ && w < we) acc[p * 3 + q] += v;
        }
      }
    }
  }
  __shared__ float red[9][8];
  const int lane = threadIdx.x & 31, wv = threadIdx.x >> 5;
#pragma unroll
  for (int j = 0; j < 9; ++j) {
    float v = acc[j];
    for (int off = 16; off > 0; off >>= 1) v += __shfl_down(v, off);
    if (lane == 0) red[j][wv] = v;
  }
  __syncthreads();
  if (threadIdx.x < 9) {
    float s = 0.f;
#pragma unroll
    for (int w2 = 0; w2 < 8; ++w2) s += red[threadIdx.x][w2];
    pooled[(size_t)bc * 9 + threadIdx.x] = s * (1.0f / 484.0f);
  }
}

// ---------------- K2: gf[b,c,:] = filter_gen_w[c,:] . pooled[b,:,:] + bias[c]
__global__ __launch_bounds__(256) void dcm_fgen(const float* __restrict__ pooled,
                                                const float* __restrict__ w,
                                                const float* __restrict__ bias,
                                                float* __restrict__ gf) {
  const int c = blockIdx.x, b = blockIdx.y;
  const float* pw = pooled + (size_t)b * CIN * 9;
  const float* wr = w + (size_t)c * CIN;
  float acc[9];
#pragma unroll
  for (int j = 0; j < 9; ++j) acc[j] = 0.f;
  for (int k = threadIdx.x; k < CIN; k += 256) {
    const float wv2 = wr[k];
    const float* pp = pw + (size_t)k * 9;
#pragma unroll
    for (int j = 0; j < 9; ++j) acc[j] += wv2 * pp[j];
  }
  __shared__ float red[9][8];
  const int lane = threadIdx.x & 31, wv = threadIdx.x >> 5;
#pragma unroll
  for (int j = 0; j < 9; ++j) {
    float v = acc[j];
    for (int off = 16; off > 0; off >>= 1) v += __shfl_down(v, off);
    if (lane == 0) red[j][wv] = v;
  }
  __syncthreads();
  if (threadIdx.x < 9) {
    float s = 0.f;
#pragma unroll
    for (int w2 = 0; w2 < 8; ++w2) s += red[threadIdx.x][w2];
    gf[((size_t)b * CMID + c) * 9 + threadIdx.x] = s + bias[c];
  }
}

// ---------------- K3/K5: Out[nb,M,Ns] = relu(W[M,K] @ In[nb,K,Ns] + bias)
// Split-f16 WMMA GEMM. Block tile 128x64, BK=32, 256 threads (8 waves).
#define BM 128
#define BN 64
#define BK 32
#define AP 40   // LDS row pitch in halfs (80B: 16B-aligned, conflict-free)
#define BP 40

__global__ __launch_bounds__(256) void dcm_gemm_relu(
    const float* __restrict__ Wm, const float* __restrict__ In,
    const float* __restrict__ bias, float* __restrict__ Out,
    int M, int K, int Ns) {
  __shared__ __align__(16) _Float16 Ah[BM * AP];
  __shared__ __align__(16) _Float16 Al[BM * AP];
  __shared__ __align__(16) _Float16 Bh[BN * BP];
  __shared__ __align__(16) _Float16 Bl[BN * BP];

  const int t    = threadIdx.x;
  const int lane = t & 31;
  const int wv   = t >> 5;
  const int m0   = blockIdx.y * BM;
  const int n0   = blockIdx.x * BN;
  const float* Inb = In + (size_t)blockIdx.z * K * Ns;

  v8f acc[4];
#pragma unroll
  for (int j = 0; j < 4; ++j)
#pragma unroll
    for (int e = 0; e < 8; ++e) acc[j][e] = 0.f;

  // --- staging assignments (per-thread constants, float4-granular) ---
  // A tile 128x32 = 1024 float4 -> 4/thread ; B tile 32x64 = 512 float4 -> 2/thread
  int amr[4], akr[4];
  const float* aptr[4];
#pragma unroll
  for (int r = 0; r < 4; ++r) {
    const int idx4 = t + (r << 8);
    amr[r] = idx4 >> 3;           // row m (0..127)
    akr[r] = (idx4 & 7) << 2;     // k quad offset (0..28)
    aptr[r] = Wm + (size_t)(m0 + amr[r]) * K + akr[r];
  }
  int bkr[2], bnr[2];
  const float* bptr[2];
#pragma unroll
  for (int r = 0; r < 2; ++r) {
    const int idx4 = t + (r << 8);
    bkr[r] = idx4 >> 4;           // k row (0..31)
    bnr[r] = (idx4 & 15) << 2;    // n quad offset (0..60)
    bptr[r] = Inb + (size_t)bkr[r] * Ns + n0 + bnr[r];
  }

  // --- preload tile 0 into registers ---
  float4 areg[4], breg[2];
#pragma unroll
  for (int r = 0; r < 4; ++r) { areg[r] = *(const float4*)aptr[r]; aptr[r] += BK; }
#pragma unroll
  for (int r = 0; r < 2; ++r) { breg[r] = *(const float4*)bptr[r]; bptr[r] += (size_t)BK * Ns; }

  // ISA 16-bit fragment lane mapping (wave32):
  // A 16x32: lane<16 -> K 0..7 & 16..23 ; lane>=16 -> K 8..15 & 24..31
  // B 32x16: lane<16 -> K 0..7 & 16..23 ; lane>=16 -> K 8..15 & 24..31
  const int am    = (wv << 4) + (lane & 15);
  const int khalf = (lane & 16) ? 8 : 0;
  const int koff  = (lane & 16) ? 16 : 0;
  const int bn    = lane & 15;

  for (int k0 = 0; k0 < K; k0 += BK) {
    // ---- convert + store staged registers to LDS (hi/lo split-f16) ----
#pragma unroll
    for (int r = 0; r < 4; ++r) {
      const float fv[4] = {areg[r].x, areg[r].y, areg[r].z, areg[r].w};
      h4 hi, lo;
#pragma unroll
      for (int e = 0; e < 4; ++e) {
        const _Float16 h = (_Float16)fv[e];
        hi[e] = h;
        lo[e] = (_Float16)(fv[e] - (float)h);
      }
      *(h4*)&Ah[amr[r] * AP + akr[r]] = hi;
      *(h4*)&Al[amr[r] * AP + akr[r]] = lo;
    }
#pragma unroll
    for (int r = 0; r < 2; ++r) {
      const float fv[4] = {breg[r].x, breg[r].y, breg[r].z, breg[r].w};
#pragma unroll
      for (int e = 0; e < 4; ++e) {
        const int n = bnr[r] + e;
        const _Float16 h = (_Float16)fv[e];
        Bh[n * BP + bkr[r]] = h;
        Bl[n * BP + bkr[r]] = (_Float16)(fv[e] - (float)h);
      }
    }
    __syncthreads();

    // ---- issue global loads for tile k+1 (overlap with WMMA below) ----
    if (k0 + BK < K) {
#pragma unroll
      for (int r = 0; r < 4; ++r) { areg[r] = *(const float4*)aptr[r]; aptr[r] += BK; }
#pragma unroll
      for (int r = 0; r < 2; ++r) { breg[r] = *(const float4*)bptr[r]; bptr[r] += (size_t)BK * Ns; }
      if (k0 + 2 * BK < K) {   // prefetch the HBM-resident B stream 2 tiles ahead
        __builtin_prefetch(bptr[0], 0, 0);
        __builtin_prefetch(bptr[1], 0, 0);
      }
    }

    // ---- fragments: two ds_read_b128 each (16B-aligned, pitch 80B) ----
    const h8 a0 = *(const h8*)(&Ah[am * AP + khalf]);
    const h8 a1 = *(const h8*)(&Ah[am * AP + 16 + khalf]);
    const v16h afh = CAT16(a0, a1);
    const h8 l0 = *(const h8*)(&Al[am * AP + khalf]);
    const h8 l1 = *(const h8*)(&Al[am * AP + 16 + khalf]);
    const v16h afl = CAT16(l0, l1);

#pragma unroll
    for (int j = 0; j < 4; ++j) {
      const int col = (j << 4) + bn;
      const h8 b0 = *(const h8*)(&Bh[col * BP + khalf]);
      const h8 b1 = *(const h8*)(&Bh[col * BP + 16 + khalf]);
      const v16h bfh = CAT16(b0, b1);
      const h8 c0 = *(const h8*)(&Bl[col * BP + khalf]);
      const h8 c1 = *(const h8*)(&Bl[col * BP + 16 + khalf]);
      const v16h bfl = CAT16(c0, c1);
      acc[j] = __builtin_amdgcn_wmma_f32_16x16x32_f16(false, afh, false, bfh,
                                                      (short)0, acc[j], false, false);
      acc[j] = __builtin_amdgcn_wmma_f32_16x16x32_f16(false, afh, false, bfl,
                                                      (short)0, acc[j], false, false);
      acc[j] = __builtin_amdgcn_wmma_f32_16x16x32_f16(false, afl, false, bfh,
                                                      (short)0, acc[j], false, false);
    }
    __syncthreads();
  }

  // C/D layout: lane<16 -> N=lane, M=r ; lane>=16 -> N=lane-16, M=r+8
  const int mrow = (wv << 4) + ((lane >> 4) << 3);
#pragma unroll
  for (int j = 0; j < 4; ++j) {
    const int ncol = n0 + (j << 4) + bn;
#pragma unroll
    for (int r = 0; r < 8; ++r) {
      const int m = m0 + mrow + r;
      float v = acc[j][r] + bias[m];
      v = v > 0.f ? v : 0.f;
      Out[((size_t)blockIdx.z * M + m) * Ns + ncol] = v;
    }
  }
}

// ---------------- K4: per-(b,c) depthwise 3x3 (zero pad) + relu
__global__ __launch_bounds__(256) void dcm_dw(const float* __restrict__ xr,
                                              const float* __restrict__ gf,
                                              float* __restrict__ out2) {
  const int c = blockIdx.y, b = blockIdx.z;
  const size_t bc = (size_t)b * CMID + c;
  const float* img = xr + bc * HW;
  const float* f = gf + bc * 9;
  float fr[9];
#pragma unroll
  for (int j = 0; j < 9; ++j) fr[j] = f[j];
  const int idx = blockIdx.x * 256 + threadIdx.x;
  const int h = idx >> 6, w = idx & 63;
  float s = 0.f;
#pragma unroll
  for (int di = -1; di <= 1; ++di) {
#pragma unroll
    for (int dj = -1; dj <= 1; ++dj) {
      const int hh = h + di, ww = w + dj;
      if (hh >= 0 && hh < 64 && ww >= 0 && ww < 64)
        s += img[hh * 64 + ww] * fr[(di + 1) * 3 + (dj + 1)];
    }
  }
  out2[bc * HW + idx] = s > 0.f ? s : 0.f;
}

// ---------------------------------------------------------------------------
extern "C" void kernel_launch(void* const* d_in, const int* in_sizes, int n_in,
                              void* d_out, int out_size, void* d_ws, size_t ws_size,
                              hipStream_t stream) {
  const float* x   = (const float*)d_in[0];   // [4,2048,64,64]
  const float* fgw = (const float*)d_in[1];   // [512,2048]
  const float* fgb = (const float*)d_in[2];   // [512]
  const float* rw  = (const float*)d_in[3];   // [512,2048]
  const float* rb  = (const float*)d_in[4];   // [512]
  const float* fw  = (const float*)d_in[5];   // [512,512]
  const float* fb  = (const float*)d_in[6];   // [512]
  float* out = (float*)d_out;                 // [4,512,64,64]

  char* ws = (char*)d_ws;
  float* xr     = (float*)ws;                                   // 32 MiB
  float* o2     = (float*)(ws + (size_t)33554432);              // 32 MiB
  float* pooled = (float*)(ws + (size_t)67108864);              // 288 KiB
  float* gf     = (float*)(ws + (size_t)67108864 + 294912);     // 72 KiB

  dcm_pool<<<NB * CIN, 256, 0, stream>>>(x, pooled);
  dcm_fgen<<<dim3(CMID, NB), 256, 0, stream>>>(pooled, fgw, fgb, gf);
  dcm_gemm_relu<<<dim3(HW / BN, CMID / BM, NB), 256, 0, stream>>>(
      rw, x, rb, xr, CMID, CIN, HW);
  dcm_dw<<<dim3(HW / 256, CMID, NB), 256, 0, stream>>>(xr, gf, o2);
  dcm_gemm_relu<<<dim3(HW / BN, CMID / BM, NB), 256, 0, stream>>>(
      fw, o2, fb, out, CMID, CMID, HW);
}
